// MaxwellsDemonGNN_71176198029758
// MI455X (gfx1250) — matrix operand order repaired and checked by reading, not compiled, not run on hardware
//
#include <hip/hip_runtime.h>
#include <hip/hip_fp16.h>

// MaxwellsDemonGNN — factored formulation for MI455X (gfx1250, wave32, WMMA).
//
// nodes = relu(x@W1+b1)@W2+b2
// Ai = nodes@We1[:H] + be1 ; Aj = nodes@We1[H:]
// Di = nodes@Wd1[:H] + bd1 ; Dj = nodes@Wd1[H:]
// per pair (i,j): hidden = relu(Ai[i]+Aj[j]); g = relu(Di[i]+Dj[j])
//                 keep = sigmoid(g.Wd2 + bd2)
// Hk[i] = sum_j hidden*keep ; Hs[i] = sum_j hidden ; K[i] = sum_j keep
// active = Hk@We2 + K*be2 ; trash = Hs@We2 - active + (S-K)*be2
// main_emb = mean_i(nodes+active); trash_emb = mean_i(trash); classifier.

typedef __attribute__((ext_vector_type(16))) _Float16 v16h;
typedef __attribute__((ext_vector_type(8)))  _Float16 v8h;
typedef __attribute__((ext_vector_type(8)))  float    v8f;

static constexpr int Bc = 4, Sc = 512, INc = 32, Hc = 64, NCc = 3;
static constexpr int Rc = Bc * Sc;   // 2048 rows

// ---------------------------------------------------------------- f32 -> f16
__global__ void cvt_f32_f16(const float* __restrict__ src,
                            _Float16* __restrict__ dst, int n) {
  int i = blockIdx.x * 256 + threadIdx.x;
  if (i < n) dst[i] = (_Float16)src[i];
}

// ------------------------------------------------- pack weight to B fragments
// src: f32 row-major [K x N]. dst layout: for n-tile t, k-chunk kc, lane l,
// element e (contiguous 16 halfs per (t,kc,l)):
//   dst[(((t*KC + kc)*32 + l)*16 + e] = src[(kc*32 + (l>>4)*16 + e)*N + t*16 + (l&15)]
// This is the contiguous k order (k = 16*(l>>4) + e) the GEMM also uses for A, so the
// GEMM kernel loads a lane's whole fragment as one contiguous 32-byte access.
__global__ void pack_b(const float* __restrict__ src, _Float16* __restrict__ dst,
                       int N, int K) {
  const int KC = K / 32;
  int idx = blockIdx.x * 256 + threadIdx.x;
  if (idx >= N * K) return;
  int tile = idx / (KC * 512);
  int rem  = idx % (KC * 512);
  int kc   = rem / 512;
  int lr   = rem % 512;
  int lane = lr / 16;
  int e    = lr % 16;
  int k = kc * 32 + (lane >> 4) * 16 + e;
  int n = tile * 16 + (lane & 15);
  dst[idx] = (_Float16)src[k * N + n];
}

// ------------------------------------------------------- WMMA GEMM (f16->f32)
// C[M,N] = A[M,K] @ B[K,N] (+bias[N]); one wave per 16x16 D tile.
// A: row-major f16 (lane fragment = two contiguous b128 loads).
// Bp: pre-packed fragments (lane fragment = one contiguous 32B load).
template <int K, bool RELU, bool SF, bool SH>
__global__ __launch_bounds__(32)
void wmma_gemm(const _Float16* __restrict__ A, const _Float16* __restrict__ Bp,
               const float* __restrict__ bias, float* __restrict__ Cf,
               _Float16* __restrict__ Ch, int N) {
  constexpr int KC = K / 32;
  const int tm = blockIdx.x, tn = blockIdx.y;
  const int lane = threadIdx.x;
  const int hsel = lane >> 4;      // lane half select
  const int l15  = lane & 15;
  const int mrow = tm * 16 + l15;  // A row carried by this lane
  const int ncol = tn * 16 + l15;  // B/D column carried by this lane

  const _Float16* __restrict__ arow = A + (size_t)mrow * K + hsel * 16;
  const _Float16* __restrict__ bfrag =
      Bp + ((size_t)tn * KC * 32 + lane) * 16;

  v8f acc = {};
#pragma unroll
  for (int kc = 0; kc < KC; ++kc) {
    // A 16x32 f16 fragment: K = k0+hsel*16 + {0..15}
    v8h alo = *(const v8h*)(arow + kc * 32);
    v8h ahi = *(const v8h*)(arow + kc * 32 + 8);
    v16h a = __builtin_shufflevector(alo, ahi, 0, 1, 2, 3, 4, 5, 6, 7,
                                     8, 9, 10, 11, 12, 13, 14, 15);
    // B 32x16 f16 fragment: contiguous 16 halfs (32B) in packed buffer
    v16h b = *(const v16h*)(bfrag + (size_t)kc * 32 * 16);
    acc = __builtin_amdgcn_wmma_f32_16x16x32_f16(
        /*neg_a=*/false, a, /*neg_b=*/false, b,
        /*c_mod=*/(short)0, acc, /*reuse_a=*/false, /*reuse_b=*/false);
  }

  const float bv = bias ? bias[ncol] : 0.f;
#pragma unroll
  for (int r = 0; r < 8; ++r) {
    int m = tm * 16 + r + hsel * 8;     // D: VGPR r -> row r (+8 for hi lanes)
    float v = acc[r] + bv;
    if (RELU) v = fmaxf(v, 0.f);
    if (SF) Cf[(size_t)m * N + ncol] = v;
    if (SH) Ch[(size_t)m * N + ncol] = (_Float16)v;
  }
}

// ----------------------------------------------------------- all-pairs kernel
// One block (8 waves) per node i. Lane k owns hidden comps {k, k+32}, gate k.
__global__ __launch_bounds__(256)
void pair_kernel(const float* __restrict__ Ai, const float* __restrict__ Aj,
                 const float* __restrict__ Di, const float* __restrict__ Dj,
                 const float* __restrict__ Wd2, const float* __restrict__ bd2,
                 float* __restrict__ Hk, float* __restrict__ Hs,
                 float* __restrict__ Ksum,
                 _Float16* __restrict__ Hkh, _Float16* __restrict__ Hsh) {
  const int i = blockIdx.x;            // global node (b*S + s)
  const int bBase = (i >> 9) << 9;     // first node of this batch
  const int lane = threadIdx.x & 31;
  const int wave = threadIdx.x >> 5;

  const float ai0 = Ai[i * Hc + lane];
  const float ai1 = Ai[i * Hc + 32 + lane];
  const float di  = Di[i * 32 + lane];
  const float w2  = Wd2[lane];
  const float bd  = bd2[0];

  float hk0 = 0.f, hk1 = 0.f, hs0 = 0.f, hs1 = 0.f, ks = 0.f;
  for (int j = wave; j < Sc; j += 8) {
    const int gj = bBase + j;
    float h0 = fmaxf(ai0 + Aj[gj * Hc + lane], 0.f);
    float h1 = fmaxf(ai1 + Aj[gj * Hc + 32 + lane], 0.f);
    float g  = fmaxf(di + Dj[gj * 32 + lane], 0.f);
    float p  = g * w2;                             // 32-wide gate dot product
    p += __shfl_xor(p, 16);
    p += __shfl_xor(p, 8);
    p += __shfl_xor(p, 4);
    p += __shfl_xor(p, 2);
    p += __shfl_xor(p, 1);
    float keep = 1.f / (1.f + __expf(-(p + bd))); // uniform across wave
    hk0 += h0 * keep; hk1 += h1 * keep;
    hs0 += h0;        hs1 += h1;
    ks  += keep;
  }

  __shared__ float sHk[8][Hc];
  __shared__ float sHs[8][Hc];
  __shared__ float sK[8];
  sHk[wave][lane] = hk0; sHk[wave][lane + 32] = hk1;
  sHs[wave][lane] = hs0; sHs[wave][lane + 32] = hs1;
  if (lane == 0) sK[wave] = ks;
  __syncthreads();

  const int t = threadIdx.x;
  if (t < Hc) {
    float a = 0.f, s = 0.f;
#pragma unroll
    for (int w = 0; w < 8; ++w) { a += sHk[w][t]; s += sHs[w][t]; }
    Hk[i * Hc + t] = a;               Hs[i * Hc + t] = s;
    Hkh[i * Hc + t] = (_Float16)a;    Hsh[i * Hc + t] = (_Float16)s;
  }
  if (t == Hc) {
    float k = 0.f;
#pragma unroll
    for (int w = 0; w < 8; ++w) k += sK[w];
    Ksum[i] = k;
  }
}

// -------------------------------------------------- per-batch pooling + head
__global__ __launch_bounds__(128)
void final_kernel(const float* __restrict__ nodes,
                  const float* __restrict__ activeRaw,   // Hk@We2
                  const float* __restrict__ sumRaw,      // Hs@We2
                  const float* __restrict__ Ksum, const float* __restrict__ be2,
                  const float* __restrict__ Wc1, const float* __restrict__ bc1,
                  const float* __restrict__ Wc2, const float* __restrict__ bc2,
                  float* __restrict__ out) {
  const int b = blockIdx.x;
  const int base = b * Sc;
  const int t = threadIdx.x;
  __shared__ float feat[2 * Hc];
  __shared__ float c1[32];

  if (t < Hc) {                       // main_emb component t
    float m = 0.f;
    for (int s = 0; s < Sc; ++s) {
      int r = base + s;
      m += nodes[r * Hc + t] + activeRaw[r * Hc + t] + Ksum[r] * be2[t];
    }
    feat[t] = m / (float)Sc;
  } else {                            // trash_emb component t-64
    int h = t - Hc;
    float tr = 0.f;
    for (int s = 0; s < Sc; ++s) {
      int r = base + s;
      tr += (sumRaw[r * Hc + h] - activeRaw[r * Hc + h]) +
            ((float)Sc - Ksum[r]) * be2[h];
    }
    feat[Hc + h] = tr / (float)Sc;
  }
  __syncthreads();

  if (t < 32) {
    float v = bc1[t];
    for (int f = 0; f < 2 * Hc; ++f) v += feat[f] * Wc1[f * 32 + t];
    c1[t] = fmaxf(v, 0.f);
  }
  __syncthreads();

  if (t < NCc) {
    float o = bc2[t];
    for (int u = 0; u < 32; ++u) o += c1[u] * Wc2[u * NCc + t];
    out[b * NCc + t] = o;
  }
}

// ---------------------------------------------------------------------- host
extern "C" void kernel_launch(void* const* d_in, const int* in_sizes, int n_in,
                              void* d_out, int out_size, void* d_ws,
                              size_t ws_size, hipStream_t stream) {
  (void)in_sizes; (void)n_in; (void)out_size; (void)ws_size;
  const float* x   = (const float*)d_in[0];
  const float* W1  = (const float*)d_in[1];
  const float* b1  = (const float*)d_in[2];
  const float* W2  = (const float*)d_in[3];
  const float* b2  = (const float*)d_in[4];
  const float* We1 = (const float*)d_in[5];
  const float* be1 = (const float*)d_in[6];
  const float* We2 = (const float*)d_in[7];
  const float* be2 = (const float*)d_in[8];
  const float* Wd1 = (const float*)d_in[9];
  const float* bd1 = (const float*)d_in[10];
  const float* Wd2 = (const float*)d_in[11];
  const float* bd2 = (const float*)d_in[12];
  const float* Wc1 = (const float*)d_in[13];
  const float* bc1 = (const float*)d_in[14];
  const float* Wc2 = (const float*)d_in[15];
  const float* bc2 = (const float*)d_in[16];
  float* out = (float*)d_out;

  char* p = (char*)d_ws;
  auto alloc = [&](size_t bytes) -> void* {
    void* r = (void*)p;
    p += (bytes + 255) & ~(size_t)255;
    return r;
  };
  _Float16* xh      = (_Float16*)alloc((size_t)Rc * INc * 2);
  _Float16* W1p     = (_Float16*)alloc((size_t)INc * Hc * 2);      // packed
  _Float16* W2p     = (_Float16*)alloc((size_t)Hc * Hc * 2);       // packed
  _Float16* We1pT   = (_Float16*)alloc((size_t)Hc * Hc * 2);       // packed top
  _Float16* We1pB   = (_Float16*)alloc((size_t)Hc * Hc * 2);       // packed bot
  _Float16* Wd1pT   = (_Float16*)alloc((size_t)Hc * 32 * 2);       // packed top
  _Float16* Wd1pB   = (_Float16*)alloc((size_t)Hc * 32 * 2);       // packed bot
  _Float16* We2p    = (_Float16*)alloc((size_t)Hc * Hc * 2);       // packed
  _Float16* th      = (_Float16*)alloc((size_t)Rc * Hc * 2);
  float*    nodes   = (float*)alloc((size_t)Rc * Hc * 4);
  _Float16* nodesh  = (_Float16*)alloc((size_t)Rc * Hc * 2);
  float*    Ai      = (float*)alloc((size_t)Rc * Hc * 4);
  float*    Aj      = (float*)alloc((size_t)Rc * Hc * 4);
  float*    Di      = (float*)alloc((size_t)Rc * 32 * 4);
  float*    Dj      = (float*)alloc((size_t)Rc * 32 * 4);
  float*    Hk      = (float*)alloc((size_t)Rc * Hc * 4);
  float*    Hs      = (float*)alloc((size_t)Rc * Hc * 4);
  _Float16* Hkh     = (_Float16*)alloc((size_t)Rc * Hc * 2);
  _Float16* Hsh     = (_Float16*)alloc((size_t)Rc * Hc * 2);
  float*    Ksum    = (float*)alloc((size_t)Rc * 4);
  float*    activeR = (float*)alloc((size_t)Rc * Hc * 4);
  float*    sumR    = (float*)alloc((size_t)Rc * Hc * 4);

  // x to f16 (A operand of first GEMM); weights packed to fragment layout.
  cvt_f32_f16<<<(Rc * INc + 255) / 256, 256, 0, stream>>>(x, xh, Rc * INc);
  auto pack = [&](const float* s, _Float16* d, int N, int K) {
    pack_b<<<(N * K + 255) / 256, 256, 0, stream>>>(s, d, N, K);
  };
  pack(W1,            W1p,  Hc, INc);   // [32x64]
  pack(W2,            W2p,  Hc, Hc);    // [64x64]
  pack(We1,           We1pT, Hc, Hc);   // We1 rows 0..63
  pack(We1 + Hc * Hc, We1pB, Hc, Hc);   // We1 rows 64..127
  pack(Wd1,           Wd1pT, 32, Hc);   // Wd1 rows 0..63
  pack(Wd1 + Hc * 32, Wd1pB, 32, Hc);   // Wd1 rows 64..127
  pack(We2,           We2p,  Hc, Hc);   // [64x64]

  const dim3 wave(32);
  const dim3 gH(Rc / 16, Hc / 16);      // N=64 tiles
  const dim3 gD(Rc / 16, 32 / 16);      // N=32 tiles
  // t = relu(x@W1 + b1)                      [2048x32]@[32x64] -> f16
  wmma_gemm<INc, true, false, true>
      <<<gH, wave, 0, stream>>>(xh, W1p, b1, nullptr, th, Hc);
  // nodes = t@W2 + b2                        [2048x64]@[64x64] -> f32 + f16
  wmma_gemm<Hc, false, true, true>
      <<<gH, wave, 0, stream>>>(th, W2p, b2, nodes, nodesh, Hc);
  // Ai = nodes@We1[:H] + be1 ; Aj = nodes@We1[H:]
  wmma_gemm<Hc, false, true, false>
      <<<gH, wave, 0, stream>>>(nodesh, We1pT, be1, Ai, nullptr, Hc);
  wmma_gemm<Hc, false, true, false>
      <<<gH, wave, 0, stream>>>(nodesh, We1pB, nullptr, Aj, nullptr, Hc);
  // Di = nodes@Wd1[:H] + bd1 ; Dj = nodes@Wd1[H:]
  wmma_gemm<Hc, false, true, false>
      <<<gD, wave, 0, stream>>>(nodesh, Wd1pT, bd1, Di, nullptr, 32);
  wmma_gemm<Hc, false, true, false>
      <<<gD, wave, 0, stream>>>(nodesh, Wd1pB, nullptr, Dj, nullptr, 32);

  // all-pairs demon gate + hidden accumulation (L2-resident, elementwise)
  pair_kernel<<<Rc, 256, 0, stream>>>(Ai, Aj, Di, Dj, Wd2, bd2,
                                      Hk, Hs, Ksum, Hkh, Hsh);

  // activeRaw = Hk@We2 ; sumRaw = Hs@We2
  wmma_gemm<Hc, false, true, false>
      <<<gH, wave, 0, stream>>>(Hkh, We2p, nullptr, activeR, nullptr, Hc);
  wmma_gemm<Hc, false, true, false>
      <<<gH, wave, 0, stream>>>(Hsh, We2p, nullptr, sumR, nullptr, Hc);

  // pooling + classifier head
  final_kernel<<<Bc, 128, 0, stream>>>(nodes, activeR, sumR, Ksum, be2,
                                       Wc1, bc1, Wc2, bc2, out);
}
